// Attention_17635135717804
// MI455X (gfx1250) — matrix operand order repaired and checked by reading, not compiled, not run on hardware
//
#include <hip/hip_runtime.h>
#include <hip/hip_bf16.h>

// ---------------------------------------------------------------------------
// Problem constants (match reference)
// ---------------------------------------------------------------------------
#define RESV   14
#define DIMV   512
#define KDV    64
#define HV     16
#define DV     256
#define DHV    4096            // H*D
#define HIDV   6144            // DH + 2*H*KD
#define BV     128
#define NV     196             // RES*RES
#define NROWS  25088           // B*N
#define NPAD   224             // 14 tiles of 16, 7 chunks of 32
#define SCALEF 0.125f          // KD^-0.5
#define EPSF   1e-5f

typedef __attribute__((ext_vector_type(16))) __bf16 bf16x16;
typedef __attribute__((ext_vector_type(8)))  __bf16 bf16x8;
typedef __attribute__((ext_vector_type(8)))  float  f32x8;

#define WMMA_BF16(a, b, c) \
  __builtin_amdgcn_wmma_f32_16x16x32_bf16(false, (a), false, (b), (short)0, (c), false, false)

// ---------------------------------------------------------------------------
// Fragment loaders (CDNA5 wave32 16x16x32 bf16 layouts per ISA 7.12.2)
// A (16xK=32): lanes 0-15 row M=lane, K in {0..7,16..23}; lanes 16-31 same
//              rows, K in {8..15,24..31}.
// B (K=32x16): lanes 0-15 col N=lane, K in {0..7,16..23}; lanes 16-31 same
//              cols, K in {8..15,24..31}.
// C/D:         VGPR j -> row j (lanes 0-15) / row j+8 (lanes 16-31),
//              col = lane & 15.
// ---------------------------------------------------------------------------
__device__ __forceinline__ bf16x16 loadA(const __bf16* rowBase, int rowStride,
                                         int kBase, int lane) {
  int r  = lane & 15;
  int kb = kBase + ((lane & 16) ? 8 : 0);
  const __bf16* p = rowBase + (size_t)r * rowStride + kb;
  bf16x8 lo = *(const bf16x8*)(p);
  bf16x8 hi = *(const bf16x8*)(p + 16);
  bf16x16 a;
#pragma unroll
  for (int i = 0; i < 8; ++i) { a[i] = lo[i]; a[i + 8] = hi[i]; }
  return a;
}

__device__ __forceinline__ bf16x16 loadA_rowclamp(const __bf16* base, int rowStride,
                                                  int row0, int maxRow, int kBase,
                                                  int lane) {
  int r = row0 + (lane & 15);
  if (r > maxRow) r = maxRow;   // padded rows: garbage but in-bounds, never stored
  int kb = kBase + ((lane & 16) ? 8 : 0);
  const __bf16* p = base + (size_t)r * rowStride + kb;
  bf16x8 lo = *(const bf16x8*)(p);
  bf16x8 hi = *(const bf16x8*)(p + 16);
  bf16x16 a;
#pragma unroll
  for (int i = 0; i < 8; ++i) { a[i] = lo[i]; a[i + 8] = hi[i]; }
  return a;
}

__device__ __forceinline__ bf16x16 loadB(const __bf16* colBase, int rowStride,
                                         int kBase, int lane) {
  int n  = lane & 15;
  int kb = kBase + ((lane & 16) ? 8 : 0);
  const __bf16* p = colBase + (size_t)n * rowStride + kb;
  bf16x8 lo = *(const bf16x8*)(p);
  bf16x8 hi = *(const bf16x8*)(p + 16);
  bf16x16 b;
#pragma unroll
  for (int i = 0; i < 8; ++i) { b[i] = lo[i]; b[i + 8] = hi[i]; }
  return b;
}

// P-fragment: read f32 probs from LDS, fold in 1/rowsum, convert to bf16.
__device__ __forceinline__ bf16x16 loadP(const float* srow, int mBase, float inv,
                                         int lane) {
  int r  = lane & 15;
  int kb = mBase + ((lane & 16) ? 8 : 0);
  const float* p = srow + r * NPAD + kb;
  bf16x16 a;
#pragma unroll
  for (int i = 0; i < 8; ++i) {
    a[i]     = (__bf16)(p[i]      * inv);
    a[i + 8] = (__bf16)(p[i + 16] * inv);
  }
  return a;
}

// ---------------------------------------------------------------------------
// Stage 0: f32 -> bf16 conversion, bias materialization
// ---------------------------------------------------------------------------
__global__ __launch_bounds__(256) void cvt_bf16_kernel(const float* __restrict__ src,
                                                       __bf16* __restrict__ dst, int n) {
  int i = blockIdx.x * 256 + threadIdx.x;
  if (i < n) dst[i] = (__bf16)src[i];
}

__global__ __launch_bounds__(256) void build_bias_kernel(const float* __restrict__ table,
                                                         const int* __restrict__ idxs,
                                                         float* __restrict__ bmat) {
  int i = blockIdx.x * 256 + threadIdx.x;
  if (i < HV * NV * NV) {
    int h = i / (NV * NV);
    int p = i - h * (NV * NV);
    bmat[i] = table[h * NV + idxs[p]];
  }
}

// ---------------------------------------------------------------------------
// Stage 1: qkv = x @ w_qkv^T, affine, scatter to per-head q/k/v (bf16)
// wave tile: 16 rows x 64 cols; grid (1568, 12), 8 waves/block
// ---------------------------------------------------------------------------
__global__ __launch_bounds__(256) void qkv_gemm_kernel(
    const __bf16* __restrict__ xb, const __bf16* __restrict__ wb,
    const float* __restrict__ g1, const float* __restrict__ b1,
    const float* __restrict__ m1, const float* __restrict__ v1,
    __bf16* __restrict__ qw, __bf16* __restrict__ kw, __bf16* __restrict__ vw) {
  int lane = threadIdx.x & 31;
  int wv   = threadIdx.x >> 5;
  int rt   = blockIdx.x;                  // row tile (16 rows of x)
  int cb64 = blockIdx.y * 8 + wv;         // 64-col group, 0..95
  const __bf16* arow = xb + (size_t)rt * 16 * DIMV;

  f32x8 acc[4];
#pragma unroll
  for (int s = 0; s < 4; ++s)
#pragma unroll
    for (int j = 0; j < 8; ++j) acc[s][j] = 0.f;

  for (int kc = 0; kc < DIMV; kc += 32) {
    bf16x16 a = loadA(arow, DIMV, kc, lane);
#pragma unroll
    for (int s = 0; s < 4; ++s) {
      bf16x16 bf = loadB(wb + (size_t)(cb64 * 64 + s * 16) * DIMV, DIMV, kc, lane);
      acc[s] = WMMA_BF16(a, bf, acc[s]);
    }
  }

  int col = lane & 15;
  int rb  = (lane & 16) ? 8 : 0;
#pragma unroll
  for (int s = 0; s < 4; ++s) {
    int c = cb64 * 64 + s * 16 + col;
    float sc = g1[c] * rsqrtf(v1[c] + EPSF);
    float mm = m1[c], bb = b1[c];
    int hh = c / 384, wi = c - hh * 384;
#pragma unroll
    for (int j = 0; j < 8; ++j) {
      int r    = rt * 16 + rb + j;
      int bidx = r / NV;
      int n    = r - bidx * NV;
      float val = (acc[s][j] - mm) * sc + bb;
      __bf16 bv = (__bf16)val;
      size_t hb = ((size_t)(bidx * HV + hh)) * NV + n;
      if (wi < KDV)            qw[hb * KDV + wi]            = bv;
      else if (wi < 2 * KDV)   kw[hb * KDV + (wi - KDV)]    = bv;
      else                     vw[hb * DV + (wi - 2 * KDV)] = bv;
    }
  }
}

// ---------------------------------------------------------------------------
// Stage 2: fused attention per (b,h). LDS: K (224x64), V^T (256x224),
// per-wave score slabs 8 x (16x224 f32), row stats. ~253 KB dynamic LDS
// (WGP supports up to 320 KB/workgroup on CDNA5).
// ---------------------------------------------------------------------------
#define K_LDS_OFF  0
#define V_LDS_OFF  (NPAD * KDV * 2)                      // 28672
#define S_LDS_OFF  (V_LDS_OFF + DV * NPAD * 2)           // 143360
#define ST_LDS_OFF (S_LDS_OFF + 8 * 16 * NPAD * 4)       // 258048
#define SMEM_TOTAL (ST_LDS_OFF + 8 * 16 * 4)             // 258560

__global__ __launch_bounds__(256) void attn_kernel(
    const __bf16* __restrict__ qw, const __bf16* __restrict__ kw,
    const __bf16* __restrict__ vw, const float* __restrict__ bias_mat,
    __bf16* __restrict__ ow) {
  extern __shared__ char smem[];
  __bf16* k_lds = (__bf16*)(smem + K_LDS_OFF);   // [224][64] row-major
  __bf16* v_lds = (__bf16*)(smem + V_LDS_OFF);   // [256][224] d-major (transposed)
  float*  s_lds = (float*)(smem + S_LDS_OFF);    // 8 waves x [16][224]
  float*  stats = (float*)(smem + ST_LDS_OFF);   // 8 waves x 16 inv-sums

  int bh = blockIdx.x;
  int b  = bh >> 4, h = bh & 15;
  int tid = threadIdx.x, lane = tid & 31, wv = tid >> 5;

  const __bf16* qg = qw + (size_t)bh * NV * KDV;
  const __bf16* kg = kw + (size_t)bh * NV * KDV;
  const __bf16* vg = vw + (size_t)bh * NV * DV;

  // stage K (pad rows >= 196 with zeros)
  for (int i = tid; i < NPAD * KDV; i += 256) {
    int n = i >> 6;
    k_lds[i] = (n < NV) ? kg[i] : (__bf16)0.0f;
  }
  // stage V transposed, 8 elems per thread iteration
  for (int i = tid; i < NPAD * 32; i += 256) {
    int n  = i >> 5;
    int d8 = (i & 31) * 8;
    bf16x8 t;
    if (n < NV) {
      t = *(const bf16x8*)(vg + (size_t)n * DV + d8);
    } else {
#pragma unroll
      for (int j = 0; j < 8; ++j) t[j] = (__bf16)0.0f;
    }
#pragma unroll
    for (int j = 0; j < 8; ++j) v_lds[(size_t)(d8 + j) * NPAD + n] = t[j];
  }
  __syncthreads();

  const float* bmat = bias_mat + (size_t)h * NV * NV;
  float* srow = s_lds + wv * (16 * NPAD);
  int col = lane & 15;
  int rb  = (lane & 16) ? 8 : 0;

  for (int qt = wv; qt < NPAD / 16; qt += 8) {
    // ---- S = q @ k^T * SCALE + bias -------------------------------------
    bf16x16 qa0 = loadA_rowclamp(qg, KDV, qt * 16, NV - 1, 0, lane);
    bf16x16 qa1 = loadA_rowclamp(qg, KDV, qt * 16, NV - 1, 32, lane);
    for (int mt = 0; mt < NPAD / 16; ++mt) {
      f32x8 c;
#pragma unroll
      for (int j = 0; j < 8; ++j) c[j] = 0.f;
      bf16x16 kb0 = loadB(k_lds + (size_t)mt * 16 * KDV, KDV, 0, lane);
      bf16x16 kb1 = loadB(k_lds + (size_t)mt * 16 * KDV, KDV, 32, lane);
      c = WMMA_BF16(qa0, kb0, c);
      c = WMMA_BF16(qa1, kb1, c);
      int mg = mt * 16 + col;
#pragma unroll
      for (int j = 0; j < 8; ++j) {
        int row = rb + j;
        int n   = qt * 16 + row;
        if (n > NV - 1) n = NV - 1;
        float s = (mg < NV) ? (c[j] * SCALEF + bmat[(size_t)n * NV + mg]) : -1e30f;
        srow[row * NPAD + mt * 16 + col] = s;
      }
    }
    // ---- softmax (each row handled by a lane pair) ----------------------
    {
      int row = lane & 15;
      int cb  = (lane & 16) ? 112 : 0;
      float* sp = srow + row * NPAD + cb;
      float mx = -1e30f;
      for (int c2 = 0; c2 < 112; ++c2) mx = fmaxf(mx, sp[c2]);
      mx = fmaxf(mx, __shfl_xor(mx, 16, 32));
      float sum = 0.f;
      for (int c2 = 0; c2 < 112; ++c2) {
        float e = __expf(sp[c2] - mx);
        sp[c2] = e;
        sum += e;
      }
      sum += __shfl_xor(sum, 16, 32);
      if (!(lane & 16)) stats[wv * 16 + row] = 1.f / (sum + 1e-30f);
    }
    // ---- O = P @ V, hardswish, store ------------------------------------
    f32x8 oacc[16];
#pragma unroll
    for (int dt = 0; dt < 16; ++dt)
#pragma unroll
      for (int j = 0; j < 8; ++j) oacc[dt][j] = 0.f;

    float inv = stats[wv * 16 + (lane & 15)];
    for (int mc = 0; mc < NPAD / 32; ++mc) {
      bf16x16 pa = loadP(srow, mc * 32, inv, lane);
#pragma unroll
      for (int dt = 0; dt < 16; ++dt) {
        bf16x16 vb = loadB(v_lds + (size_t)dt * 16 * NPAD, NPAD, mc * 32, lane);
        oacc[dt] = WMMA_BF16(pa, vb, oacc[dt]);
      }
    }
#pragma unroll
    for (int dt = 0; dt < 16; ++dt) {
#pragma unroll
      for (int j = 0; j < 8; ++j) {
        int row = rb + j;
        int n   = qt * 16 + row;
        if (n < NV) {
          float x = oacc[dt][j];
          float t = x + 3.f;
          t = (t < 0.f) ? 0.f : (t > 6.f ? 6.f : t);
          float o = x * t * (1.f / 6.f);
          ow[((size_t)(b * NV + n)) * DHV + h * DV + dt * 16 + col] = (__bf16)o;
        }
      }
    }
  }
}

// ---------------------------------------------------------------------------
// Stage 3: y = o @ w_proj^T, affine, fp32 out. wave tile 16x64; 8 waves = 512
// ---------------------------------------------------------------------------
__global__ __launch_bounds__(256) void proj_gemm_kernel(
    const __bf16* __restrict__ ob, const __bf16* __restrict__ wb,
    const float* __restrict__ g2, const float* __restrict__ b2,
    const float* __restrict__ m2, const float* __restrict__ v2,
    float* __restrict__ out) {
  int lane = threadIdx.x & 31;
  int wv   = threadIdx.x >> 5;
  int rt   = blockIdx.x;
  int cb64 = wv;                     // 8 waves x 64 cols = 512
  const __bf16* arow = ob + (size_t)rt * 16 * DHV;

  f32x8 acc[4];
#pragma unroll
  for (int s = 0; s < 4; ++s)
#pragma unroll
    for (int j = 0; j < 8; ++j) acc[s][j] = 0.f;

  for (int kc = 0; kc < DHV; kc += 32) {
    bf16x16 a = loadA(arow, DHV, kc, lane);
#pragma unroll
    for (int s = 0; s < 4; ++s) {
      bf16x16 bf = loadB(wb + (size_t)(cb64 * 64 + s * 16) * DHV, DHV, kc, lane);
      acc[s] = WMMA_BF16(a, bf, acc[s]);
    }
  }

  int col = lane & 15;
  int rb  = (lane & 16) ? 8 : 0;
#pragma unroll
  for (int s = 0; s < 4; ++s) {
    int c = cb64 * 64 + s * 16 + col;
    float sc = g2[c] * rsqrtf(v2[c] + EPSF);
    float mm = m2[c], bb = b2[c];
#pragma unroll
    for (int j = 0; j < 8; ++j) {
      int r = rt * 16 + rb + j;
      out[(size_t)r * DIMV + c] = (acc[s][j] - mm) * sc + bb;
    }
  }
}

// ---------------------------------------------------------------------------
// Workspace layout (bytes, all 256-aligned); total ~527 MB
// ---------------------------------------------------------------------------
static const size_t OFF_XBF   = 0;
static const size_t OFF_WQKV  = OFF_XBF   + (size_t)NROWS * DIMV * 2;
static const size_t OFF_WPROJ = OFF_WQKV  + (size_t)HIDV * DIMV * 2;
static const size_t OFF_BIAS  = OFF_WPROJ + (size_t)DIMV * DHV * 2;
static const size_t OFF_QW    = OFF_BIAS  + (size_t)HV * NV * NV * 4;
static const size_t OFF_KW    = OFF_QW    + (size_t)BV * HV * NV * KDV * 2;
static const size_t OFF_VW    = OFF_KW    + (size_t)BV * HV * NV * KDV * 2;
static const size_t OFF_OW    = OFF_VW    + (size_t)BV * HV * NV * DV * 2;

extern "C" void kernel_launch(void* const* d_in, const int* in_sizes, int n_in,
                              void* d_out, int out_size, void* d_ws, size_t ws_size,
                              hipStream_t stream) {
  (void)in_sizes; (void)n_in; (void)out_size; (void)ws_size;
  const float* x          = (const float*)d_in[0];
  const float* w_qkv      = (const float*)d_in[1];
  const float* g1         = (const float*)d_in[2];
  const float* b1         = (const float*)d_in[3];
  const float* m1         = (const float*)d_in[4];
  const float* v1         = (const float*)d_in[5];
  const float* bias_table = (const float*)d_in[6];
  const float* w_proj     = (const float*)d_in[7];
  const float* g2         = (const float*)d_in[8];
  const float* b2         = (const float*)d_in[9];
  const float* m2         = (const float*)d_in[10];
  const float* v2         = (const float*)d_in[11];
  const int*   bias_idxs  = (const int*)d_in[12];
  float* out = (float*)d_out;

  char* ws = (char*)d_ws;
  __bf16* xb     = (__bf16*)(ws + OFF_XBF);
  __bf16* wqkvb  = (__bf16*)(ws + OFF_WQKV);
  __bf16* wprojb = (__bf16*)(ws + OFF_WPROJ);
  float*  bmat   = (float*)(ws + OFF_BIAS);
  __bf16* qw     = (__bf16*)(ws + OFF_QW);
  __bf16* kw     = (__bf16*)(ws + OFF_KW);
  __bf16* vw     = (__bf16*)(ws + OFF_VW);
  __bf16* ow     = (__bf16*)(ws + OFF_OW);

  // Stage 0: precision conversion + bias materialization
  {
    int n0 = NROWS * DIMV;
    cvt_bf16_kernel<<<(n0 + 255) / 256, 256, 0, stream>>>(x, xb, n0);
    int n1 = HIDV * DIMV;
    cvt_bf16_kernel<<<(n1 + 255) / 256, 256, 0, stream>>>(w_qkv, wqkvb, n1);
    int n2 = DIMV * DHV;
    cvt_bf16_kernel<<<(n2 + 255) / 256, 256, 0, stream>>>(w_proj, wprojb, n2);
    int n3 = HV * NV * NV;
    build_bias_kernel<<<(n3 + 255) / 256, 256, 0, stream>>>(bias_table, bias_idxs, bmat);
  }
  // Stage 1: QKV projection + affine -> q/k/v (bf16, per-head layout)
  qkv_gemm_kernel<<<dim3(NROWS / 16, HIDV / 64 / 8), 256, 0, stream>>>(
      xb, wqkvb, g1, b1, m1, v1, qw, kw, vw);
  // Stage 2: fused attention (one workgroup per (b,h))
  attn_kernel<<<BV * HV, 256, SMEM_TOTAL, stream>>>(qw, kw, vw, bmat, ow);
  // Stage 3: output projection + affine -> fp32 out
  proj_gemm_kernel<<<NROWS / 16, 256, 0, stream>>>(ow, wprojb, g2, b2, m2, v2, out);
}
